// LSTMCell_49349174231639
// MI455X (gfx1250) — matrix operand order repaired and checked by reading, not compiled, not run on hardware
//
#include <hip/hip_runtime.h>

// ---------------------------------------------------------------------------
// LSTM cell for gfx1250 (MI455X): wave32 bf16 WMMA + async global->LDS fills.
//   Stage 1 (pack):  X,h0 -> bf16 ; 8 weights -> bf16 transposed W^T[N][K]
//   Stage 2: cand = X@W_hi + h0@W_hh + b          (fp32 + bf16 out)
//   Stage 3: gates i/o/f share A tiles; fused sigmoid/tanh cell epilogue.
// ---------------------------------------------------------------------------

#define BDIM 4096
#define KDIM 1024
#define NDIM 1024

typedef __attribute__((ext_vector_type(16))) __bf16 v16bf;
typedef __attribute__((ext_vector_type(8)))  float  v8f;

struct alignas(16) F4   { float x, y, z, w; };
struct alignas(8)  BF4  { unsigned short x, y, z, w; };
struct alignas(16) U128 { unsigned int a, b, c, d; };
union  FragU { U128 q[2]; v16bf v; };

// fp32 -> bf16 round-to-nearest-even (bit pattern)
__device__ __forceinline__ unsigned short f2bf(float f) {
    unsigned u = __builtin_bit_cast(unsigned, f);
    u = (u + 0x7FFFu + ((u >> 16) & 1u)) >> 16;
    return (unsigned short)u;
}

__device__ __forceinline__ float sigmoidf(float x) {
    return 1.0f / (1.0f + __expf(-x));
}

// CDNA5 async copy: 16 bytes global -> LDS per lane, tracked by ASYNCcnt.
__device__ __forceinline__ void async_b128(unsigned lds_off, const void* gptr) {
    asm volatile("global_load_async_to_lds_b128 %0, %1, off"
                 :: "v"(lds_off), "v"(gptr) : "memory");
}
__device__ __forceinline__ void wait_async_le(int n) {
    asm volatile("s_wait_asynccnt %0" :: "i"(n) : "memory");
}

// LDS A tile [128 rows][40 bf16] (80-B row stride). ISA wave32 A-frag (16x32):
// lane half h, row r -> 16B chunks at K=8h and K=16+8h.
__device__ __forceinline__ v16bf load_fragA(const unsigned short* lds,
                                            int row, int half) {
    const char* p = (const char*)lds + 80 * row + 16 * half;
    FragU u;
    u.q[0] = *(const U128*)(p);
    u.q[1] = *(const U128*)(p + 32);
    return u.v;
}
// LDS B tile (W^T) [64 N-rows][40 bf16]. ISA B-frag (32x16): lane col n,
// half h -> 16B chunks at K=8h and K=16+8h (same order as A).
__device__ __forceinline__ v16bf load_fragB(const unsigned short* lds,
                                            int n, int half) {
    const char* p = (const char*)lds + 80 * n + 16 * half;
    FragU u;
    u.q[0] = *(const U128*)(p);
    u.q[1] = *(const U128*)(p + 32);
    return u.v;
}

// --------------------------- pack kernels ----------------------------------
__global__ __launch_bounds__(256)
void pack_act(const float* __restrict__ src, unsigned short* __restrict__ dst) {
    int i = blockIdx.x * 256 + threadIdx.x;         // one float4 per thread
    F4 v = ((const F4*)src)[i];
    BF4 p = { f2bf(v.x), f2bf(v.y), f2bf(v.z), f2bf(v.w) };
    ((BF4*)dst)[i] = p;
}

struct WPack { const float* w[8]; unsigned short* wt[8]; };

// Transpose W[K][N] fp32 -> W^T[N][K] bf16 via 64x64 LDS tile.
__global__ __launch_bounds__(256)
void pack_wT(WPack p) {
    __shared__ unsigned short t[64 * 66];
    const float* W = p.w[blockIdx.z];
    unsigned short* WT = p.wt[blockIdx.z];
    const int kb = blockIdx.y * 64, nb = blockIdx.x * 64;
    for (int u = threadIdx.x; u < 4096; u += 256) {
        int k = u >> 6, n = u & 63;                 // coalesced along n
        t[n * 66 + k] = f2bf(W[(size_t)(kb + k) * NDIM + nb + n]);
    }
    __syncthreads();
    for (int u = threadIdx.x; u < 1024; u += 256) {
        int n = u >> 4, k4 = (u & 15) << 2;         // coalesced along k
        BF4 q = { t[n * 66 + k4], t[n * 66 + k4 + 1],
                  t[n * 66 + k4 + 2], t[n * 66 + k4 + 3] };
        *(BF4*)&WT[(size_t)(nb + n) * KDIM + kb + k4] = q;
    }
}

// --------------------------- GEMM kernels ----------------------------------
// WG = 128(M) x 64(N), 8 waves as 4x2, each wave 32x32 (4 WMMA tiles / gate).

__global__ __launch_bounds__(256)
void lstm_cand_gemm(const unsigned short* __restrict__ Xbf,
                    const unsigned short* __restrict__ H0bf,
                    const unsigned short* __restrict__ WThi,
                    const unsigned short* __restrict__ WThh,
                    const float* __restrict__ bhi, const float* __restrict__ bhh,
                    float* __restrict__ cand32,
                    unsigned short* __restrict__ candbf) {
    __shared__ unsigned short lds[2 * (128 * 40 + 64 * 40)];
    const unsigned ldsbase = (unsigned)(unsigned long long)(void*)lds;

    const int tid  = threadIdx.x;
    const int wid  = tid >> 5, lane = tid & 31;
    const int half = lane >> 4, lr = lane & 15;
    const int mb   = blockIdx.y * 128, nb = blockIdx.x * 64;
    const int wm   = (wid >> 1) * 32,  wn = (wid & 1) * 32;
    constexpr int ABYTES = 128 * 40 * 2;            // 10240
    constexpr int BUFW   = 128 * 40 + 64 * 40;      // ushorts per buffer
    constexpr int BUFB   = BUFW * 2;                // 15360 bytes

    auto issue = [&](int buf, int kk) {
        const unsigned short* A = (kk < KDIM) ? Xbf : H0bf;
        const unsigned short* W = (kk < KDIM) ? WThi : WThh;
        const int kb = kk & (KDIM - 1);
        const unsigned base = ldsbase + buf * BUFB;
        #pragma unroll
        for (int u = 0; u < 2; ++u) {               // A: 512 16B chunks
            int idx = tid + u * 256;
            int m = idx >> 2, c = idx & 3;
            async_b128(base + m * 80 + c * 16,
                       A + (size_t)(mb + m) * KDIM + kb + c * 8);
        }
        {                                           // B: 256 16B chunks
            int n = tid >> 2, c = tid & 3;
            async_b128(base + ABYTES + n * 80 + c * 16,
                       W + (size_t)(nb + n) * KDIM + kb + c * 8);
        }
    };

    v8f acc[2][2] = {};
    issue(0, 0);
    #pragma unroll 1
    for (int it = 0; it < 64; ++it) {
        const int nxt = it + 1;
        if (nxt < 64) issue(nxt & 1, nxt * 32);
        if (nxt < 64) wait_async_le(3); else wait_async_le(0);
        __syncthreads();

        const unsigned short* bufA = lds + (it & 1) * BUFW;
        const unsigned short* bufB = bufA + 128 * 40;
        v16bf af[2], bf[2];
        #pragma unroll
        for (int i = 0; i < 2; ++i) af[i] = load_fragA(bufA, wm + i * 16 + lr, half);
        #pragma unroll
        for (int i = 0; i < 2; ++i) bf[i] = load_fragB(bufB, wn + i * 16 + lr, half);
        #pragma unroll
        for (int mi = 0; mi < 2; ++mi)
            #pragma unroll
            for (int ni = 0; ni < 2; ++ni)
                acc[mi][ni] = __builtin_amdgcn_wmma_f32_16x16x32_bf16(
                    false, af[mi], false, bf[ni], (short)0, acc[mi][ni],
                    false, false);
        __syncthreads();
    }

    #pragma unroll
    for (int mi = 0; mi < 2; ++mi)
        #pragma unroll
        for (int ni = 0; ni < 2; ++ni) {
            int n = nb + wn + ni * 16 + lr;
            float bias = bhi[n] + bhh[n];
            #pragma unroll
            for (int j = 0; j < 8; ++j) {
                int m = mb + wm + mi * 16 + half * 8 + j;
                size_t idx = (size_t)m * NDIM + n;
                float v = acc[mi][ni][j] + bias;
                cand32[idx] = v;
                candbf[idx] = f2bf(v);
            }
        }
}

__global__ __launch_bounds__(256)
void lstm_gates_gemm(const unsigned short* __restrict__ Xbf,
                     const unsigned short* __restrict__ candbf,
                     const float* __restrict__ cand32,
                     const float* __restrict__ C0,
                     const unsigned short* __restrict__ WTii,
                     const unsigned short* __restrict__ WTih,
                     const unsigned short* __restrict__ WToi,
                     const unsigned short* __restrict__ WToh,
                     const unsigned short* __restrict__ WTfi,
                     const unsigned short* __restrict__ WTfh,
                     const float* __restrict__ bii, const float* __restrict__ bih,
                     const float* __restrict__ boi, const float* __restrict__ boh,
                     const float* __restrict__ bfi, const float* __restrict__ bfh,
                     float* __restrict__ outH, float* __restrict__ outC) {
    __shared__ unsigned short lds[2 * (128 * 40 + 3 * 64 * 40)];
    const unsigned ldsbase = (unsigned)(unsigned long long)(void*)lds;

    const int tid  = threadIdx.x;
    const int wid  = tid >> 5, lane = tid & 31;
    const int half = lane >> 4, lr = lane & 15;
    const int mb   = blockIdx.y * 128, nb = blockIdx.x * 64;
    const int wm   = (wid >> 1) * 32,  wn = (wid & 1) * 32;
    constexpr int ABYTES = 128 * 40 * 2;            // 10240
    constexpr int GBYTES = 64 * 40 * 2;             // 5120 per gate
    constexpr int BUFW   = 128 * 40 + 3 * 64 * 40;  // ushorts per buffer
    constexpr int BUFB   = BUFW * 2;                // 25600 bytes

    auto issue = [&](int buf, int kk) {
        const unsigned short* A = (kk < KDIM) ? Xbf : candbf;
        const unsigned short* Wg[3];
        Wg[0] = (kk < KDIM) ? WTii : WTih;
        Wg[1] = (kk < KDIM) ? WToi : WToh;
        Wg[2] = (kk < KDIM) ? WTfi : WTfh;
        const int kb = kk & (KDIM - 1);
        const unsigned base = ldsbase + buf * BUFB;
        #pragma unroll
        for (int u = 0; u < 2; ++u) {
            int idx = tid + u * 256;
            int m = idx >> 2, c = idx & 3;
            async_b128(base + m * 80 + c * 16,
                       A + (size_t)(mb + m) * KDIM + kb + c * 8);
        }
        const int n = tid >> 2, c = tid & 3;
        #pragma unroll
        for (int g = 0; g < 3; ++g)
            async_b128(base + ABYTES + g * GBYTES + n * 80 + c * 16,
                       Wg[g] + (size_t)(nb + n) * KDIM + kb + c * 8);
    };

    v8f acc[3][2][2] = {};
    issue(0, 0);
    #pragma unroll 1
    for (int it = 0; it < 64; ++it) {
        const int nxt = it + 1;
        if (nxt < 64) issue(nxt & 1, nxt * 32);
        if (nxt < 64) wait_async_le(5); else wait_async_le(0);
        __syncthreads();

        const unsigned short* bufA = lds + (it & 1) * BUFW;
        v16bf af[2];
        #pragma unroll
        for (int i = 0; i < 2; ++i) af[i] = load_fragA(bufA, wm + i * 16 + lr, half);
        #pragma unroll
        for (int g = 0; g < 3; ++g) {
            const unsigned short* bufB = bufA + 128 * 40 + g * 64 * 40;
            v16bf bf[2];
            #pragma unroll
            for (int i = 0; i < 2; ++i) bf[i] = load_fragB(bufB, wn + i * 16 + lr, half);
            #pragma unroll
            for (int mi = 0; mi < 2; ++mi)
                #pragma unroll
                for (int ni = 0; ni < 2; ++ni)
                    acc[g][mi][ni] = __builtin_amdgcn_wmma_f32_16x16x32_bf16(
                        false, af[mi], false, bf[ni], (short)0, acc[g][mi][ni],
                        false, false);
        }
        __syncthreads();
    }

    #pragma unroll
    for (int mi = 0; mi < 2; ++mi)
        #pragma unroll
        for (int ni = 0; ni < 2; ++ni) {
            int n = nb + wn + ni * 16 + lr;
            float bi  = bii[n] + bih[n];
            float bo  = boi[n] + boh[n];
            float bff = bfi[n] + bfh[n];
            #pragma unroll
            for (int j = 0; j < 8; ++j) {
                int m = mb + wm + mi * 16 + half * 8 + j;
                size_t idx = (size_t)m * NDIM + n;
                float iv = sigmoidf(acc[0][mi][ni][j] + bi);
                float ov = sigmoidf(acc[1][mi][ni][j] + bo);
                float fv = sigmoidf(acc[2][mi][ni][j] + bff);
                float cd = cand32[idx];
                float ct = fv * C0[idx] + iv * tanhf(cd);
                outH[idx] = tanhf(ct) * ov;
                outC[idx] = ct;
            }
        }
}

// ----------------------------- launcher ------------------------------------
extern "C" void kernel_launch(void* const* d_in, const int* in_sizes, int n_in,
                              void* d_out, int out_size, void* d_ws, size_t ws_size,
                              hipStream_t stream) {
    const float* X   = (const float*)d_in[0];
    const float* H0  = (const float*)d_in[1];
    const float* C0  = (const float*)d_in[2];
    const float* bhi = (const float*)d_in[4];
    const float* bhh = (const float*)d_in[6];
    const float* bii = (const float*)d_in[8];
    const float* bih = (const float*)d_in[10];
    const float* boi = (const float*)d_in[12];
    const float* boh = (const float*)d_in[14];
    const float* bfi = (const float*)d_in[16];
    const float* bfh = (const float*)d_in[18];

    float* outH = (float*)d_out;
    float* outC = (float*)d_out + (size_t)BDIM * NDIM;

    // workspace layout (bytes):
    //   cand32: 16 MB | candbf: 8 MB | Xbf: 8 MB | H0bf: 8 MB | WT[8]: 16 MB
    char* ws = (char*)d_ws;
    float*          cand32 = (float*)ws;                         ws += (size_t)BDIM * NDIM * 4;
    unsigned short* candbf = (unsigned short*)ws;                ws += (size_t)BDIM * NDIM * 2;
    unsigned short* Xbf    = (unsigned short*)ws;                ws += (size_t)BDIM * KDIM * 2;
    unsigned short* H0bf   = (unsigned short*)ws;                ws += (size_t)BDIM * KDIM * 2;
    unsigned short* WT[8];
    for (int i = 0; i < 8; ++i) { WT[i] = (unsigned short*)ws;   ws += (size_t)KDIM * NDIM * 2; }

    WPack wp;
    for (int i = 0; i < 8; ++i) { wp.w[i] = (const float*)d_in[3 + 2 * i]; wp.wt[i] = WT[i]; }

    dim3 blk(256);
    pack_act<<<dim3((BDIM * KDIM / 4) / 256), blk, 0, stream>>>(X, Xbf);
    pack_act<<<dim3((BDIM * KDIM / 4) / 256), blk, 0, stream>>>(H0, H0bf);
    pack_wT<<<dim3(NDIM / 64, KDIM / 64, 8), blk, 0, stream>>>(wp);

    dim3 grid(NDIM / 64, BDIM / 128);
    lstm_cand_gemm<<<grid, blk, 0, stream>>>(Xbf, H0bf, WT[0], WT[1],
                                             bhi, bhh, cand32, candbf);
    lstm_gates_gemm<<<grid, blk, 0, stream>>>(Xbf, candbf, cand32, C0,
                                              WT[2], WT[3], WT[4], WT[5], WT[6], WT[7],
                                              bii, bih, boi, boh, bfi, bfh,
                                              outH, outC);
}
